// DistModule_41008347742480
// MI455X (gfx1250) — matrix operand order repaired and checked by reading, not compiled, not run on hardware
//
#include <hip/hip_runtime.h>
#include <hip/hip_bf16.h>

typedef __attribute__((ext_vector_type(16))) _Float16 v16h;
typedef __attribute__((ext_vector_type(8)))  float    v8f;

#define C_DIM 256
#define HW    16384        // 128*128 pixels per image
#define NPIX  65536        // B * HW
#define EPSF  1e-5f
#define SLOPE 0.01f

// ---------------------------------------------------------------------------
// k0: convert W1 (f32 -> f16) and zero the global stat accumulators
// ---------------------------------------------------------------------------
__global__ __launch_bounds__(256) void k0_prep(const float* __restrict__ W1,
                                               _Float16* __restrict__ w1h,
                                               float* __restrict__ gsum,
                                               float* __restrict__ gsq) {
    int i = blockIdx.x * 256 + threadIdx.x;      // 0 .. 65535
    w1h[i] = (_Float16)W1[i];
    if (i < C_DIM) { gsum[i] = 0.0f; gsq[i] = 0.0f; }
}

// ---------------------------------------------------------------------------
// k1: feat1_t[b][p][c] = sum_c' W1[c][c'] * x[b][c'][p] + b1[c]
// f16 WMMA 16x16x32, f32 accumulate. Block tile 64(o) x 128(p), 8 waves,
// each wave computes a 32x32 region via 2x2 WMMA tiles, K-loop of 8 steps.
// B tile is staged in LDS PRE-SWIZZLED into the WMMA B-fragment layout so
// each lane's v16h fragment is one contiguous 32B read (2x ds_load_b128).
// ---------------------------------------------------------------------------
__global__ __launch_bounds__(256) void k1_gemm(const float* __restrict__ x,
                                               const _Float16* __restrict__ w1h,
                                               const float* __restrict__ b1,
                                               float* __restrict__ feat1t) {
    __shared__ _Float16 As[64][40];                          // 64 o-rows x 32 k (pad 8)
    __shared__ __attribute__((aligned(64))) _Float16 BsF[8][32][24]; // [p-sub][lane][kpos], 48B lane stride

    const int tid  = threadIdx.x;
    const int b    = blockIdx.z;
    const int o0   = blockIdx.y * 64;
    const int p0   = blockIdx.x * 128;
    const int wv   = tid >> 5;
    const int lane = tid & 31;
    const int wo   = wv >> 2;          // 0..1 : o sub-block (32 rows)
    const int wp   = wv & 3;           // 0..3 : p sub-block (32 cols)
    const int M    = lane & 15;
    const int hf   = lane >> 4;

    // B staging decomposition: 256 thr = 16 k-row-pairs x 8 p-subtiles x 2 pixel-halves
    const int rp  = tid >> 4;          // 0..15 : rows 2rp, 2rp+1
    const int rem = tid & 15;
    const int seg = rem >> 1;          // 0..7  : p subtile
    const int hp  = rem & 1;           // 0..1  : which 8 pixels of the subtile
    const int laneBase = ((rp >> 3) << 4) + hp * 8;   // (k>>4)*16 + n_base
    const int pos      = (rp & 7) << 1;               // k & 15 (even)

    v8f acc[2][2] = {};

    for (int kb = 0; kb < C_DIM; kb += 32) {
        __syncthreads();
        // --- stage A: W1h rows [o0, o0+64), cols [kb, kb+32)  (f16, 16B ops)
        {
            int row = tid >> 2, cg = tid & 3;
            const uint4* src = (const uint4*)(w1h + (size_t)(o0 + row) * C_DIM + kb + cg * 8);
            *(uint4*)&As[row][cg * 8] = *src;
        }
        // --- stage B swizzled: rows kb+2rp, kb+2rp+1; 8 pixels; pack half2 per pixel
        {
            const float* s0 = x + ((size_t)(b * C_DIM + kb + 2 * rp)) * HW + p0 + seg * 16 + hp * 8;
            float4 a0 = ((const float4*)s0)[0];
            float4 a1 = ((const float4*)s0)[1];
            float4 c0 = ((const float4*)(s0 + HW))[0];
            float4 c1 = ((const float4*)(s0 + HW))[1];
            float va[8] = {a0.x,a0.y,a0.z,a0.w,a1.x,a1.y,a1.z,a1.w};
            float vb[8] = {c0.x,c0.y,c0.z,c0.w,c1.x,c1.y,c1.z,c1.w};
            #pragma unroll
            for (int i = 0; i < 8; ++i) {
                union { _Float16 h[2]; unsigned int u; } pk;
                pk.h[0] = (_Float16)va[i];
                pk.h[1] = (_Float16)vb[i];
                *(unsigned int*)&BsF[seg][laneBase + i][pos] = pk.u;
            }
        }
        __syncthreads();

        // --- A fragments: contiguous halves -> 2x ds_load_b128 each
        v16h afr[2], bfr[2];
        #pragma unroll
        for (int t = 0; t < 2; ++t) {
            int r = wo * 32 + t * 16 + M;
            #pragma unroll
            for (int v = 0; v < 8; ++v) {
                int kk = v * 2 + hf * 16;
                afr[t][2 * v]     = As[r][kk];
                afr[t][2 * v + 1] = As[r][kk + 1];
            }
        }
        // --- B fragments: pre-swizzled, one 32B contiguous read per fragment
        #pragma unroll
        for (int s = 0; s < 2; ++s) {
            int sub = wp * 2 + s;
            bfr[s] = *(const v16h*)&BsF[sub][lane][0];
        }
        #pragma unroll
        for (int t = 0; t < 2; ++t)
            #pragma unroll
            for (int s = 0; s < 2; ++s)
                acc[t][s] = __builtin_amdgcn_wmma_f32_16x16x32_f16(
                    false, afr[t], false, bfr[s], (short)0, acc[t][s], false, false);
    }

    // --- store transposed: feat1_t[b][p][c]; lane holds 8 consecutive channels
    #pragma unroll
    for (int t = 0; t < 2; ++t) {
        int c = o0 + wo * 32 + t * 16 + hf * 8;
        float4 bb0 = *(const float4*)(b1 + c);
        float4 bb1 = *(const float4*)(b1 + c + 4);
        #pragma unroll
        for (int s = 0; s < 2; ++s) {
            int p = p0 + wp * 32 + s * 16 + M;
            float* dst = feat1t + ((size_t)(b * HW + p)) * C_DIM + c;
            float4 lo, hi;
            lo.x = acc[t][s][0] + bb0.x; lo.y = acc[t][s][1] + bb0.y;
            lo.z = acc[t][s][2] + bb0.z; lo.w = acc[t][s][3] + bb0.w;
            hi.x = acc[t][s][4] + bb1.x; hi.y = acc[t][s][5] + bb1.y;
            hi.z = acc[t][s][6] + bb1.z; hi.w = acc[t][s][7] + bb1.w;
            ((float4*)dst)[0] = lo;
            ((float4*)dst)[1] = hi;
        }
    }
}

// ---------------------------------------------------------------------------
// k2: per-pixel 8-neighbor L1-dist softmax aggregation + residual + BN stats.
// One wave per pixel, 8 pixels per wave; lane l owns channels [8l, 8l+8).
// ---------------------------------------------------------------------------
__global__ __launch_bounds__(256) void k2_agg(const float* __restrict__ feat1t,
                                              const float* __restrict__ x,
                                              const float* __restrict__ rp,
                                              float* __restrict__ out,
                                              float* __restrict__ gsum,
                                              float* __restrict__ gsq) {
    __shared__ float ssum[C_DIM];
    __shared__ float ssq[C_DIM];
    const int tid  = threadIdx.x;
    ssum[tid] = 0.0f; ssq[tid] = 0.0f;
    __syncthreads();

    const int wv    = tid >> 5;
    const int lane  = tid & 31;
    const int cbase = lane * 8;
    const float r   = rp[0];
    const int  gw   = blockIdx.x * 8 + wv;      // 0..8191

    const int dyv[8] = {-1,-1,-1, 0, 0, 1, 1, 1};
    const int dxv[8] = {-1, 0, 1,-1, 1,-1, 0, 1};

    float lsum[8] = {0,0,0,0,0,0,0,0};
    float lsq[8]  = {0,0,0,0,0,0,0,0};

    for (int it = 0; it < 8; ++it) {
        const int pg = gw + it * 8192;          // global pixel 0..65535
        const int b  = pg >> 14;
        const int p  = pg & 16383;
        const int h  = p >> 7;
        const int w  = p & 127;

        // center feature (8 channels per lane, contiguous)
        float cf[8];
        {
            const float4* cp = (const float4*)(feat1t + (size_t)pg * C_DIM + cbase);
            float4 c0 = cp[0], c1 = cp[1];
            cf[0]=c0.x; cf[1]=c0.y; cf[2]=c0.z; cf[3]=c0.w;
            cf[4]=c1.x; cf[5]=c1.y; cf[6]=c1.z; cf[7]=c1.w;
        }

        float nfv[8][8];
        float d[8];
        #pragma unroll
        for (int k = 0; k < 8; ++k) {
            int hy = h + dyv[k]; hy = hy < 0 ? 0 : (hy > 127 ? 127 : hy);
            int wx = w + dxv[k]; wx = wx < 0 ? 0 : (wx > 127 ? 127 : wx);
            int npg = (b << 14) + (hy << 7) + wx;
            const float4* np = (const float4*)(feat1t + (size_t)npg * C_DIM + cbase);
            float4 n0 = np[0], n1 = np[1];
            nfv[k][0]=n0.x; nfv[k][1]=n0.y; nfv[k][2]=n0.z; nfv[k][3]=n0.w;
            nfv[k][4]=n1.x; nfv[k][5]=n1.y; nfv[k][6]=n1.z; nfv[k][7]=n1.w;
            float s = 0.0f;
            #pragma unroll
            for (int j = 0; j < 8; ++j) s += fabsf(cf[j] - nfv[k][j]);
            d[k] = s;
        }

        // wave-wide butterfly reduction: each lane ends with full channel sum
        #pragma unroll
        for (int k = 0; k < 8; ++k) {
            #pragma unroll
            for (int off = 16; off >= 1; off >>= 1)
                d[k] += __shfl_xor(d[k], off, 32);
        }

        // softmax over the 8 neighbors (identical in all lanes)
        float mx = -r * d[0];
        #pragma unroll
        for (int k = 1; k < 8; ++k) mx = fmaxf(mx, -r * d[k]);
        float e[8], es = 0.0f;
        #pragma unroll
        for (int k = 0; k < 8; ++k) { e[k] = __expf(-r * d[k] - mx); es += e[k]; }
        const float inv = 1.0f / es;

        // weighted aggregation + residual + store + stats
        float ov[8] = {0,0,0,0,0,0,0,0};
        #pragma unroll
        for (int k = 0; k < 8; ++k) {
            float m = e[k] * inv;
            #pragma unroll
            for (int j = 0; j < 8; ++j) ov[j] = fmaf(m, nfv[k][j], ov[j]);
        }
        const size_t xb = ((size_t)(b * C_DIM + cbase)) * HW + p;
        #pragma unroll
        for (int j = 0; j < 8; ++j) {
            float v = ov[j] + x[xb + (size_t)j * HW];
            out[xb + (size_t)j * HW] = v;
            lsum[j] += v;
            lsq[j]  += v * v;
        }
    }

    #pragma unroll
    for (int j = 0; j < 8; ++j) {
        atomicAdd(&ssum[cbase + j], lsum[j]);
        atomicAdd(&ssq[cbase + j],  lsq[j]);
    }
    __syncthreads();
    atomicAdd(&gsum[tid], ssum[tid]);
    atomicAdd(&gsq[tid],  ssq[tid]);
}

// ---------------------------------------------------------------------------
// k3: fold stats into per-channel scale/shift
// ---------------------------------------------------------------------------
__global__ void k3_stats(const float* __restrict__ gsum, const float* __restrict__ gsq,
                         const float* __restrict__ gamma, const float* __restrict__ beta,
                         float* __restrict__ scale, float* __restrict__ shift) {
    int c = threadIdx.x;
    const float invN = 1.0f / (float)NPIX;
    float mean = gsum[c] * invN;
    float var  = gsq[c] * invN - mean * mean;
    float sc   = gamma[c] * rsqrtf(var + EPSF);
    scale[c] = sc;
    shift[c] = beta[c] - mean * sc;
}

// ---------------------------------------------------------------------------
// k4: in-place BN affine + LeakyReLU, float4 vectorized
// ---------------------------------------------------------------------------
__global__ __launch_bounds__(256) void k4_bn(float* __restrict__ out,
                                             const float* __restrict__ scale,
                                             const float* __restrict__ shift) {
    size_t idx  = (size_t)blockIdx.x * 256 + threadIdx.x;  // float4 index
    size_t base = idx * 4;
    int c = (int)((base >> 14) & 255);                      // innermost dim = HW
    float sc = scale[c], sh = shift[c];
    float4 v = ((float4*)out)[idx];
    float a = fmaf(v.x, sc, sh), b = fmaf(v.y, sc, sh);
    float cc = fmaf(v.z, sc, sh), dd = fmaf(v.w, sc, sh);
    v.x = a  > 0.0f ? a  : SLOPE * a;
    v.y = b  > 0.0f ? b  : SLOPE * b;
    v.z = cc > 0.0f ? cc : SLOPE * cc;
    v.w = dd > 0.0f ? dd : SLOPE * dd;
    ((float4*)out)[idx] = v;
}

// ---------------------------------------------------------------------------
extern "C" void kernel_launch(void* const* d_in, const int* in_sizes, int n_in,
                              void* d_out, int out_size, void* d_ws, size_t ws_size,
                              hipStream_t stream) {
    const float* x     = (const float*)d_in[0];
    const float* W1    = (const float*)d_in[1];
    const float* b1    = (const float*)d_in[2];
    const float* r     = (const float*)d_in[3];
    const float* gamma = (const float*)d_in[4];
    const float* beta  = (const float*)d_in[5];
    float* out = (float*)d_out;

    char* ws = (char*)d_ws;
    float*    feat1t = (float*)ws;                              // 64 MB
    _Float16* w1h    = (_Float16*)(ws + 67108864ull);           // 128 KB
    float*    gsum   = (float*)(ws + 67239936ull);              // 256 f
    float*    gsq    = gsum + 256;
    float*    scale  = gsum + 512;
    float*    shift  = gsum + 768;

    k0_prep<<<256, 256, 0, stream>>>(W1, w1h, gsum, gsq);
    k1_gemm<<<dim3(128, 4, 4), 256, 0, stream>>>(x, w1h, b1, feat1t);
    k2_agg<<<1024, 256, 0, stream>>>(feat1t, x, r, out, gsum, gsq);
    k3_stats<<<1, 256, 0, stream>>>(gsum, gsq, gamma, beta, scale, shift);
    k4_bn<<<16384, 256, 0, stream>>>(out, scale, shift);
}
